// GlobalAttention_hype_52432960749613
// MI455X (gfx1250) — matrix operand order repaired and checked by reading, not compiled, not run on hardware
//
#include <hip/hip_runtime.h>

#define DEVI __device__ __forceinline__

typedef __attribute__((ext_vector_type(16))) __bf16 v16bf;
typedef __attribute__((ext_vector_type(8)))  __bf16 v8bf;
typedef __attribute__((ext_vector_type(8)))  float  v8f;
typedef __attribute__((ext_vector_type(4)))  unsigned v4u;
typedef __attribute__((ext_vector_type(8)))  int    v8i;
typedef __attribute__((ext_vector_type(4)))  int    v4i;

constexpr int Bc = 8, Tc = 2048, Sc = 2048, Dc = 128;
constexpr int LMR = 144;                 // lam*mem rows: 128 data + lam-1 + pad
constexpr float EPSc = 1e-9f;
constexpr float MAXN = 1.0f - 1e-5f;

DEVI float artanhf_(float x) {
    x = fminf(fmaxf(x, -1.0f + 1e-7f), 1.0f - 1e-7f);
    return 0.5f * (log1pf(x) - log1pf(-x));
}

DEVI v8f wmma_bf16(v16bf a, v16bf b, v8f c) {
    return __builtin_amdgcn_wmma_f32_16x16x32_bf16(false, a, false, b, (short)0, c, false, false);
}

// ---- WMMA fragment loaders (layouts per ISA 7.12.2) ------------------------
DEVI v16bf afrag_bf16(const __bf16* base, int ld, int row0, int k0) {
    int l = threadIdx.x & 31, half = l >> 4, r = l & 15;
    const __bf16* p = base + (size_t)(row0 + r) * ld + k0 + 8 * half;
    v8bf lo = *(const v8bf*)p;
    v8bf hi = *(const v8bf*)(p + 16);
    return __builtin_shufflevector(lo, hi, 0, 1, 2, 3, 4, 5, 6, 7, 8, 9, 10, 11, 12, 13, 14, 15);
}

DEVI v16bf afrag_f32(const float* base, int ld, int row0, int k0) {
    int l = threadIdx.x & 31, half = l >> 4, r = l & 15;
    const float* p = base + (size_t)(row0 + r) * ld + k0 + 8 * half;
    v16bf f;
#pragma unroll
    for (int e = 0; e < 8; e++) f[e] = (__bf16)p[e];
#pragma unroll
    for (int e = 0; e < 8; e++) f[8 + e] = (__bf16)p[16 + e];
    return f;
}

// B fragment from transposed source Bt[n][k]: lane (half,n): k = 8*half + e (e<8), 16 + 8*half + (e-8) (e>=8).
DEVI v16bf bfragT(const __bf16* bt, int ldk, int col0, int k0) {
    int l = threadIdx.x & 31, half = l >> 4, n = l & 15;
    const __bf16* p = bt + (size_t)(col0 + n) * ldk + k0 + 8 * half;
    v8bf lo = *(const v8bf*)p;
    v8bf hi = *(const v8bf*)(p + 16);
    return __builtin_shufflevector(lo, hi, 0, 1, 2, 3, 4, 5, 6, 7, 8, 9, 10, 11, 12, 13, 14, 15);
}

// ---- Tensor Data Mover: 2D bf16 tile (global -> LDS), D# per ISA 8.3/8.4 ----
#if __has_builtin(__builtin_amdgcn_tensor_load_to_lds)
#define HAVE_TDM 1
#else
#define HAVE_TDM 0
#endif

DEVI void tdm_load_bf16(unsigned lds_off, const void* gptr,
                        unsigned tile_d0, unsigned tile_d1,
                        unsigned tensor_d0, unsigned long long stride0) {
#if HAVE_TDM
    unsigned long long ga = (unsigned long long)(size_t)gptr;
    v4u g0;
    g0[0] = 1u;                                          // count=1, user desc
    g0[1] = lds_off;                                     // LDS byte address
    g0[2] = (unsigned)(ga & 0xffffffffu);                // global_addr[31:0]
    g0[3] = (unsigned)((ga >> 32) & 0x01ffffffu) | (2u << 30);  // addr[56:32] | type=2
    v8i g1;
    g1[0] = (int)(1u << 16);                             // data_size=1 (2 bytes)
    g1[1] = (int)((tensor_d0 & 0xffffu) << 16);          // tensor_dim0 lo16 @63:48
    g1[2] = (int)((tensor_d0 >> 16) | ((tile_d1 & 0xffffu) << 16)); // dim0 hi | tensor_dim1 lo
    g1[3] = (int)(((tile_d1 & 0xffff0000u) >> 16) | (tile_d0 << 16)); // dim1 hi | tile_dim0
    g1[4] = (int)(tile_d1 & 0xffffu);                    // tile_dim1 (0 => 1D)
    g1[5] = (int)(unsigned)(stride0 & 0xffffffffu);      // tensor_dim0_stride lo32
    g1[6] = (int)(unsigned)((stride0 >> 32) & 0xffffu);  // stride hi16 (dim1_stride=0)
    g1[7] = 0;
    v4i z4 = {0, 0, 0, 0};
    v8i z8 = {0, 0, 0, 0, 0, 0, 0, 0};
    __builtin_amdgcn_tensor_load_to_lds(g0, g1, z4, z4, z8, 0);
#else
    (void)lds_off; (void)gptr; (void)tile_d0; (void)tile_d1;
    (void)tensor_d0; (void)stride0;
#endif
}

DEVI void tdm_wait(int pending) {
#if HAVE_TDM
    if (pending) __builtin_amdgcn_s_wait_tensorcnt(1);
    else         __builtin_amdgcn_s_wait_tensorcnt(0);
#else
    (void)pending;
#endif
}

// ---------------- GEMM: C(rows x 128) = X(rows x K) @ W(K x 128) -------------
template <int K>
__global__ void k_gemm(const float* __restrict__ X, const float* __restrict__ W,
                       float* __restrict__ C) {
    __shared__ __bf16 Wt[Dc * K];   // transposed: Wt[n*K + k] = W[k*128 + n]
    {
        int n = threadIdx.x;
        for (int k = 0; k < K; k++) Wt[(size_t)n * K + k] = (__bf16)W[(size_t)k * Dc + n];
    }
    __syncthreads();

    int wave = threadIdx.x >> 5;
    int row0 = blockIdx.x * 64 + wave * 16;
    v8f acc[8] = {};
    for (int kk = 0; kk < K; kk += 32) {
        v16bf a = afrag_f32(X, K, row0, kk);
#pragma unroll
        for (int nt = 0; nt < 8; nt++) {
            v16bf bf = bfragT(Wt, K, nt * 16, kk);
            acc[nt] = wmma_bf16(a, bf, acc[nt]);
        }
    }
    int l = threadIdx.x & 31, half = l >> 4, n = l & 15;
#pragma unroll
    for (int nt = 0; nt < 8; nt++)
#pragma unroll
        for (int v = 0; v < 8; v++) {
            int m = v + 8 * half;
            C[(size_t)(row0 + m) * Dc + nt * 16 + n] = acc[nt][v];
        }
}

// -------------- mobius_linear row epilogue (128 wide) ------------------------
__global__ void k_rowmath(const float* __restrict__ X, const float* __restrict__ MX,
                          const float* __restrict__ bias,
                          __bf16* __restrict__ outbf, float* __restrict__ outn2) {
    int row = blockIdx.x;
    int d = threadIdx.x;
    __shared__ float red[128];
    auto bred = [&](float v) -> float {
        red[d] = v; __syncthreads();
        for (int s = 64; s > 0; s >>= 1) { if (d < s) red[d] += red[d + s]; __syncthreads(); }
        float r = red[0]; __syncthreads(); return r;
    };
    float x  = X[(size_t)row * Dc + d];
    float mx = MX[(size_t)row * Dc + d];
    float bv = bias[d];

    float x2  = bred(x * x);
    float mx2 = bred(mx * mx);
    float xn  = sqrtf(fmaxf(x2, EPSc * EPSc));
    float mxn = sqrtf(fmaxf(mx2, EPSc * EPSc));
    float res = tanhf(mxn / xn * artanhf_(xn)) / mxn * mx;
    float rn  = sqrtf(fmaxf(bred(res * res), EPSc * EPSc));
    res *= fminf(1.0f, MAXN / rn);
    float bn = sqrtf(fmaxf(bred(bv * bv), EPSc * EPSc));
    float eb = tanhf(bn) / bn * bv;
    float en = sqrtf(fmaxf(bred(eb * eb), EPSc * EPSc));
    eb *= fminf(1.0f, MAXN / en);
    float y2  = bred(eb * eb);
    float rx2 = bred(res * res);
    float xy  = bred(res * eb);
    float num = (1.f + 2.f * xy + y2) * res + (1.f - rx2) * eb;
    float den = 1.f + 2.f * xy + rx2 * y2;
    float o = num / fmaxf(den, EPSc);
    float on = sqrtf(fmaxf(bred(o * o), EPSc * EPSc));
    o *= fminf(1.0f, MAXN / on);
    float fn2 = bred(o * o);

    outbf[(size_t)row * Dc + d] = (__bf16)o;
    if (d == 0) outn2[row] = fn2;
}

// ---- lam*mem transposed to (B, 144, S); row 128 = lam-1, 129..143 = 0 -------
__global__ void k_lammem(const float* __restrict__ mem, __bf16* __restrict__ lmT) {
    int row = blockIdx.x;                // b*S + s
    int b = row / Sc, s = row % Sc;
    int d = threadIdx.x;
    __shared__ float red[128];
    float m = mem[(size_t)row * Dc + d];
    red[d] = m * m; __syncthreads();
    for (int t = 64; t > 0; t >>= 1) { if (d < t) red[d] += red[d + t]; __syncthreads(); }
    float y2 = red[0];
    float lam = 2.0f / fmaxf(1.0f - y2, EPSc);
    lmT[((size_t)b * LMR + d) * Sc + s] = (__bf16)(lam * m);
    if (d == 0) lmT[((size_t)b * LMR + 128) * Sc + s] = (__bf16)(lam - 1.0f);
    if (d >= 1 && d < 16) lmT[((size_t)b * LMR + 128 + d) * Sc + s] = (__bf16)0.0f;
}

// ---------------- scores: -dist(q_t, u_s)*point, masked, raw -----------------
// 4 waves share b; u tile (64 x 128 bf16) staged in LDS via TDM, double-buffered.
__global__ void k_score(const __bf16* __restrict__ qbf, const __bf16* __restrict__ ubf,
                        const float* __restrict__ q2, const float* __restrict__ u2,
                        const int* __restrict__ lens, const float* __restrict__ pointp,
                        float* __restrict__ attn) {
    __shared__ __bf16 uS[2][64 * Dc];    // 2 x 16KB
    int wave = threadIdx.x >> 5;
    int gt = blockIdx.x * 4 + wave;      // tile index over (B*T)/16
    int b  = gt / (Tc / 16);
    int t0 = (gt % (Tc / 16)) * 16;
    int l = threadIdx.x & 31, half = l >> 4, n = l & 15;
    float point = pointp[0];
    int len = lens[b];
    const __bf16* ub = ubf + (size_t)b * Sc * Dc;

    v16bf afr[4];
#pragma unroll
    for (int kk = 0; kk < 4; kk++) afr[kk] = afrag_bf16(qbf + (size_t)b * Tc * Dc, Dc, t0, kk * 32);
    float x2v[8];
#pragma unroll
    for (int v = 0; v < 8; v++) x2v[v] = q2[(size_t)b * Tc + t0 + v + 8 * half];

    int sbeg = blockIdx.y * (Sc / 4);
    constexpr int NST = (Sc / 4) / 64;   // 8 stages of 64 s-rows

    auto issue = [&](int buf, int s0) {
#if HAVE_TDM
        if (threadIdx.x < 32)
            tdm_load_bf16((unsigned)(size_t)&uS[buf][0], ub + (size_t)s0 * Dc,
                          /*tile_d0=*/64 * Dc, /*tile_d1=*/0,
                          /*tensor_d0=*/64 * Dc, /*stride0=*/64 * Dc);
#else
        const v8bf* src = (const v8bf*)(ub + (size_t)s0 * Dc);
        v8bf* dst = (v8bf*)&uS[buf][0];
        for (int i = 0; i < 8; i++) dst[threadIdx.x + i * 128] = src[threadIdx.x + i * 128];
#endif
    };

    issue(0, sbeg);
    for (int st = 0; st < NST; st++) {
        int s0 = sbeg + st * 64;
        if (st + 1 < NST) issue((st + 1) & 1, s0 + 64);
#if HAVE_TDM
        if (threadIdx.x < 32) tdm_wait(st + 1 < NST);
#endif
        __syncthreads();
        const __bf16* ut = &uS[st & 1][0];
#pragma unroll
        for (int sub = 0; sub < 4; sub++) {
            v8f acc = {};
#pragma unroll
            for (int kk = 0; kk < 4; kk++) {
                v16bf bf = bfragT(ut, Dc, sub * 16, kk * 32);
                acc = wmma_bf16(afr[kk], bf, acc);
            }
            int s = s0 + sub * 16 + n;
            float y2 = u2[(size_t)b * Sc + s];
            bool inr = s < len;
#pragma unroll
            for (int v = 0; v < 8; v++) {
                float xy = acc[v];
                float x2 = x2v[v];
                float sq = fmaxf(x2 + y2 - 2.0f * xy, 0.0f);
                float dn = fmaxf((1.0f - x2) * (1.0f - y2), EPSc);
                float z = 1.0f + 2.0f * sq / dn;
                float dd = logf(z + sqrtf(fmaxf(z * z - 1.0f, 0.0f)));
                float al = inr ? (-dd * point) : -1e9f;
                int t = t0 + v + 8 * half;
                attn[((size_t)t * Bc + b) * Sc + s] = al;
            }
        }
        __syncthreads();
    }
}

// ---------------------------- softmax over S, in place -----------------------
__global__ void k_softmax(float* __restrict__ attn) {
    int row = blockIdx.x;                // t*B + b
    float* p = attn + (size_t)row * Sc;
    int tid = threadIdx.x;
    __shared__ float red[256];
    float vals[8]; float mx = -1e30f;
#pragma unroll
    for (int i = 0; i < 8; i++) { vals[i] = p[tid + i * 256]; mx = fmaxf(mx, vals[i]); }
    red[tid] = mx; __syncthreads();
    for (int s = 128; s > 0; s >>= 1) { if (tid < s) red[tid] = fmaxf(red[tid], red[tid + s]); __syncthreads(); }
    mx = red[0]; __syncthreads();
    float sum = 0.f;
#pragma unroll
    for (int i = 0; i < 8; i++) { vals[i] = expf(vals[i] - mx); sum += vals[i]; }
    red[tid] = sum; __syncthreads();
    for (int s = 128; s > 0; s >>= 1) { if (tid < s) red[tid] += red[tid + s]; __syncthreads(); }
    float inv = 1.0f / red[0];
#pragma unroll
    for (int i = 0; i < 8; i++) p[tid + i * 256] = vals[i] * inv;
}

// ---- num = attn @ (lam*mem), den folded in as 9th WMMA column tile ----------
// B tile ((144 x 32) bf16) staged in LDS via TDM, double-buffered.
__global__ void k_midpoint(const float* __restrict__ attn, const __bf16* __restrict__ lmT,
                           float* __restrict__ num, float* __restrict__ den) {
    __shared__ __bf16 mS[2][LMR * 32];   // 2 x 9KB
    int wave = threadIdx.x >> 5;
    int gt = blockIdx.x * 4 + wave;
    int b  = gt / (Tc / 16);
    int t0 = (gt % (Tc / 16)) * 16;
    int l = threadIdx.x & 31, half = l >> 4, r = l & 15;
    const float* arow = attn + ((size_t)(t0 + r) * Bc + b) * Sc;
    const __bf16* lb = lmT + (size_t)b * LMR * Sc;

    auto issue = [&](int buf, int s0) {
#if HAVE_TDM
        if (threadIdx.x < 32)
            tdm_load_bf16((unsigned)(size_t)&mS[buf][0], lb + s0,
                          /*tile_d0=*/32, /*tile_d1=*/LMR,
                          /*tensor_d0=*/32, /*stride0=*/(unsigned long long)Sc);
#else
        // 144 rows x 32 bf16 = 576 chunks of 16B
        for (int j = 0; j < 5; j++) {
            int c = threadIdx.x + j * 128;
            if (c < LMR * 4) {
                int row = c >> 2, off = (c & 3) * 8;
                *(v8bf*)&mS[buf][row * 32 + off] = *(const v8bf*)(lb + (size_t)row * Sc + s0 + off);
            }
        }
#endif
    };

    v8f acc[9] = {};
    issue(0, 0);
    for (int st = 0; st < Sc / 32; st++) {
        int s0 = st * 32;
        if (st + 1 < Sc / 32) issue((st + 1) & 1, s0 + 32);
#if HAVE_TDM
        if (threadIdx.x < 32) tdm_wait(st + 1 < Sc / 32);
#endif
        __syncthreads();
        const float* p = arow + s0 + 8 * half;
        v16bf a;
#pragma unroll
        for (int e = 0; e < 8; e++) a[e] = (__bf16)p[e];
#pragma unroll
        for (int e = 0; e < 8; e++) a[8 + e] = (__bf16)p[16 + e];
        const __bf16* mt = &mS[st & 1][0];
#pragma unroll
        for (int nt = 0; nt < 9; nt++) {
            v16bf bf = bfragT(mt, 32, nt * 16, 0);
            acc[nt] = wmma_bf16(a, bf, acc[nt]);
        }
        __syncthreads();
    }
    int n = r;
#pragma unroll
    for (int nt = 0; nt < 8; nt++)
#pragma unroll
        for (int v = 0; v < 8; v++) {
            int m = v + 8 * half;
            num[((size_t)b * Tc + t0 + m) * Dc + nt * 16 + n] = acc[nt][v];
        }
    if (n == 0) {                        // lanes 0 and 16: column 128 = lam-1 dot
#pragma unroll
        for (int v = 0; v < 8; v++) {
            int m = v + 8 * half;
            den[(size_t)b * Tc + t0 + m] = acc[8][v];
        }
    }
}

// ------------- c = mobius-scale-0.5(num/den), projected ----------------------
__global__ void k_midfinish(const float* __restrict__ num, const float* __restrict__ den,
                            float* __restrict__ cbuf) {
    int row = blockIdx.x;
    int d = threadIdx.x;
    __shared__ float red[128];
    auto bred = [&](float v) -> float {
        red[d] = v; __syncthreads();
        for (int s = 64; s > 0; s >>= 1) { if (d < s) red[d] += red[d + s]; __syncthreads(); }
        float r = red[0]; __syncthreads(); return r;
    };
    float dv = den[row];
    if (fabsf(dv) < 1e-10f) dv = 1e-10f;
    float tm = num[(size_t)row * Dc + d] / dv;
    float n = sqrtf(fmaxf(bred(tm * tm), EPSc * EPSc));
    float c = tanhf(0.5f * artanhf_(fminf(n, MAXN))) / n * tm;
    float cn = sqrtf(fmaxf(bred(c * c), EPSc * EPSc));
    c *= fminf(1.0f, MAXN / cn);
    cbuf[(size_t)row * Dc + d] = c;
}

// --------------- concat [c, source] + project, store ||.||^2 -----------------
__global__ void k_concat(const float* __restrict__ cbuf, const float* __restrict__ src,
                         float* __restrict__ concat, float* __restrict__ cn2) {
    int row = blockIdx.x;
    int d = threadIdx.x;                 // 256
    __shared__ float red[256];
    float v = (d < Dc) ? cbuf[(size_t)row * Dc + d] : src[(size_t)row * Dc + (d - Dc)];
    red[d] = v * v; __syncthreads();
    for (int s = 128; s > 0; s >>= 1) { if (d < s) red[d] += red[d + s]; __syncthreads(); }
    float n = sqrtf(fmaxf(red[0], EPSc * EPSc));
    __syncthreads();
    v *= fminf(1.0f, MAXN / n);
    concat[(size_t)row * 2 * Dc + d] = v;
    red[d] = v * v; __syncthreads();
    for (int s = 128; s > 0; s >>= 1) { if (d < s) red[d] += red[d + s]; __syncthreads(); }
    if (d == 0) cn2[row] = red[0];
}

// ------ final: mobius_linear epilogue -> logmap -> LN -> tanh -> expmap ------
__global__ void k_final(const float* __restrict__ MX, const float* __restrict__ cn2,
                        const float* __restrict__ bout, const float* __restrict__ lng,
                        const float* __restrict__ lnb, float* __restrict__ out) {
    int row = blockIdx.x;                // b*T + t
    int b = row / Tc, t = row % Tc;
    int d = threadIdx.x;
    __shared__ float red[128];
    auto bred = [&](float v) -> float {
        red[d] = v; __syncthreads();
        for (int s = 64; s > 0; s >>= 1) { if (d < s) red[d] += red[d + s]; __syncthreads(); }
        float r = red[0]; __syncthreads(); return r;
    };
    float mx = MX[(size_t)row * Dc + d];
    float x2 = cn2[row];
    float mx2 = bred(mx * mx);
    float xn = sqrtf(fmaxf(x2, EPSc * EPSc));
    float mxn = sqrtf(fmaxf(mx2, EPSc * EPSc));
    float res = tanhf(mxn / xn * artanhf_(xn)) / mxn * mx;
    float rn = sqrtf(fmaxf(bred(res * res), EPSc * EPSc));
    res *= fminf(1.0f, MAXN / rn);
    float bv = bout[d];
    float bn = sqrtf(fmaxf(bred(bv * bv), EPSc * EPSc));
    float eb = tanhf(bn) / bn * bv;
    float en = sqrtf(fmaxf(bred(eb * eb), EPSc * EPSc));
    eb *= fminf(1.0f, MAXN / en);
    float y2  = bred(eb * eb);
    float rx2 = bred(res * res);
    float xy  = bred(res * eb);
    float numv = (1.f + 2.f * xy + y2) * res + (1.f - rx2) * eb;
    float denv = 1.f + 2.f * xy + rx2 * y2;
    float o = numv / fmaxf(denv, EPSc);
    float on = sqrtf(fmaxf(bred(o * o), EPSc * EPSc));
    o *= fminf(1.0f, MAXN / on);
    float lnn = sqrtf(fmaxf(bred(o * o), EPSc * EPSc));
    float y = artanhf_(lnn) / lnn * o;
    float mu = bred(y) * (1.0f / 128.0f);
    float dv0 = y - mu;
    float var = bred(dv0 * dv0) * (1.0f / 128.0f);
    float nrm = dv0 * rsqrtf(var + 1e-5f) * lng[d] + lnb[d];
    float th = tanhf(nrm);
    float tn = sqrtf(fmaxf(bred(th * th), EPSc * EPSc));
    float e = tanhf(tn) / tn * th;
    float enn = sqrtf(fmaxf(bred(e * e), EPSc * EPSc));
    e *= fminf(1.0f, MAXN / enn);
    out[((size_t)t * Bc + b) * Dc + d] = e;
}

extern "C" void kernel_launch(void* const* d_in, const int* in_sizes, int n_in,
                              void* d_out, int out_size, void* d_ws, size_t ws_size,
                              hipStream_t stream) {
    (void)in_sizes; (void)n_in; (void)out_size; (void)ws_size;
    const float* source = (const float*)d_in[0];
    const float* memory = (const float*)d_in[1];
    const int*   lens   = (const int*)d_in[2];
    const float* w_q    = (const float*)d_in[3];
    const float* b_q    = (const float*)d_in[4];
    const float* w_c    = (const float*)d_in[5];
    const float* b_c    = (const float*)d_in[6];
    const float* w_out  = (const float*)d_in[7];
    const float* b_out  = (const float*)d_in[8];
    const float* ln_g   = (const float*)d_in[9];
    const float* ln_b   = (const float*)d_in[10];
    const float* point  = (const float*)d_in[11];

    float* out    = (float*)d_out;
    float* attn_h = out;                                   // (T,B,D)
    float* attn   = out + (size_t)Tc * Bc * Dc;            // (T,B,S)

    char* ws = (char*)d_ws;
    size_t off = 0;
    auto alloc = [&](size_t bytes) -> char* {
        char* p = ws + off;
        off = (off + bytes + 255) & ~(size_t)255;
        return p;
    };
    __bf16* qbf   = (__bf16*)alloc((size_t)Bc * Tc * Dc * 2);
    __bf16* ubf   = (__bf16*)alloc((size_t)Bc * Sc * Dc * 2);
    __bf16* lmT   = (__bf16*)alloc((size_t)Bc * LMR * Sc * 2);
    float*  q2    = (float*)alloc((size_t)Bc * Tc * 4);
    float*  u2    = (float*)alloc((size_t)Bc * Sc * 4);
    float*  mxbuf = (float*)alloc((size_t)Bc * Tc * Dc * 4);
    float*  numb  = (float*)alloc((size_t)Bc * Tc * Dc * 4);
    float*  denb  = (float*)alloc((size_t)Bc * Tc * 4);
    float*  cbuf  = (float*)alloc((size_t)Bc * Tc * Dc * 4);
    float*  ccat  = (float*)alloc((size_t)Bc * Tc * 2 * Dc * 4);
    float*  cn2   = (float*)alloc((size_t)Bc * Tc * 4);
    float*  oraw  = (float*)alloc((size_t)Bc * Tc * Dc * 4);

    int rowsT = Bc * Tc;   // 16384
    int rowsS = Bc * Sc;   // 16384

    // 1) wq = mobius_linear(source, w_q, b_q)
    k_gemm<128><<<rowsT / 64, 128, 0, stream>>>(source, w_q, mxbuf);
    k_rowmath<<<rowsT, 128, 0, stream>>>(source, mxbuf, b_q, qbf, q2);
    // 2) uh = mobius_linear(memory, w_c, b_c)
    k_gemm<128><<<rowsS / 64, 128, 0, stream>>>(memory, w_c, mxbuf);
    k_rowmath<<<rowsS, 128, 0, stream>>>(memory, mxbuf, b_c, ubf, u2);
    // 3) lam*mem (transposed, 144 rows incl lam-1 column block)
    k_lammem<<<rowsS, 128, 0, stream>>>(memory, lmT);
    // 4) raw masked scores straight into attn output region (TDM-staged B)
    k_score<<<dim3((rowsT / 16) / 4, 4), 128, 0, stream>>>(qbf, ubf, q2, u2, lens, point, attn);
    // 5) softmax in place
    k_softmax<<<Tc * Bc, 256, 0, stream>>>(attn);
    // 6) weighted midpoint num + den in one WMMA pass (TDM-staged B)
    k_midpoint<<<(rowsT / 16) / 4, 128, 0, stream>>>(attn, lmT, numb, denb);
    k_midfinish<<<rowsT, 128, 0, stream>>>(numb, denb, cbuf);
    // 7) concat + project, final mobius_linear + LN chain
    k_concat<<<rowsT, 256, 0, stream>>>(cbuf, source, ccat, cn2);
    k_gemm<256><<<rowsT / 64, 128, 0, stream>>>(ccat, w_out, oraw);
    k_final<<<rowsT, 128, 0, stream>>>(oraw, cn2, b_out, ln_g, ln_b, attn_h);
}
